// TrajectoryTransformer_40853728919639
// MI455X (gfx1250) — matrix operand order repaired and checked by reading, not compiled, not run on hardware
//
#include <hip/hip_runtime.h>

// ---------------- CDNA5 (gfx1250, wave32) transformer encoder ----------------
// All GEMMs + attention use v_wmma_f32_16x16x32_bf16 (f32 accumulate).
// GEMM: 128x128 tile, double-buffered LDS, transposed W tile so both A and B
// fragments are contiguous ds_load_b128s. Attention: flash, async K/V->LDS.

typedef __attribute__((ext_vector_type(16))) __bf16 v16bf;
typedef __attribute__((ext_vector_type(8)))  float  v8f;
typedef __attribute__((ext_vector_type(4)))  int    v4i;

#define AS_GLOBAL __attribute__((address_space(1)))
#define AS_LDS    __attribute__((address_space(3)))

#if __has_builtin(__builtin_amdgcn_global_load_async_to_lds_b128)
#define HAVE_ASYNC_LDS 1
#endif

__device__ __forceinline__ void wait_asynccnt0() {
#if __has_builtin(__builtin_amdgcn_s_wait_asynccnt)
  __builtin_amdgcn_s_wait_asynccnt(0);
#else
  asm volatile("s_wait_asynccnt 0" ::: "memory");
#endif
}

namespace cfg {
constexpr int B  = 16;
constexpr int T  = 2048;
constexpr int D  = 512;
constexpr int H  = 8;
constexpr int Dh = 64;
constexpr int FF = 2048;
constexpr int BT = B * T;
constexpr int CTX = 64;
}

enum { EPI_PE = 0, EPI_QKV = 1, EPI_RELU = 2, EPI_RES = 3 };

// ---------------------------------------------------------------------------
// Generic bf16 WMMA GEMM:  C[M,N] = A[M,K] @ W[K,N] (+ bias / PE / residual)
// 256 threads = 8 waves; tile 128x128; wave tile 32x64 (2x4 WMMAs / k-step).
// ---------------------------------------------------------------------------
template <int EPI, typename AT>
__global__ __launch_bounds__(256) void gemm_bf16_kernel(
    const AT* __restrict__ A, const float* __restrict__ W,
    const float* __restrict__ bias, const float* __restrict__ resid,
    float* __restrict__ Cf, __bf16* __restrict__ Cb,
    int M, int N, int K)
{
  constexpr int LDA  = 40;  // A tile rows [m][k], 32 k + pad (bf16 elems)
  constexpr int LDBT = 40;  // W tile TRANSPOSED rows [n][k], 32 k + pad
  __shared__ __bf16 sA [2][128 * LDA];   // 2 * 10240 B
  __shared__ __bf16 sBt[2][128 * LDBT];  // 2 * 10240 B  (40 KB total)

  const int tid  = threadIdx.x;
  const int wv   = tid >> 5;
  const int lane = tid & 31;
  const int rsel = lane & 15;
  const int hi   = lane >> 4;
  const int m0 = blockIdx.y * 128;
  const int n0 = blockIdx.x * 128;
  const int wr = (wv >> 1) * 32;   // wave row offset within tile
  const int wc = (wv & 1) * 64;    // wave col offset within tile

  auto loadA = [&](int k0, int buf) {
    const int r = tid >> 1, c0 = (tid & 1) * 16;
    const AT* src = A + (size_t)(m0 + r) * K + k0 + c0;
    #pragma unroll
    for (int c = 0; c < 16; c++) sA[buf][r * LDA + c0 + c] = (__bf16)(float)src[c];
    if (k0 + 32 < K) __builtin_prefetch(src + 32, 0, 0);   // global_prefetch_b8
  };
  auto loadB = [&](int k0, int buf) {
    const int r = tid >> 3, c0 = (tid & 7) * 16;           // r = k-row, c0 = n
    const float* src = W + (size_t)(k0 + r) * N + n0 + c0;
    #pragma unroll
    for (int c = 0; c < 16; c++) sBt[buf][(c0 + c) * LDBT + r] = (__bf16)src[c];
    if (k0 + 32 < K) __builtin_prefetch(src + (size_t)32 * N, 0, 0);
  };

  v8f acc[2][4] = {};

  loadA(0, 0);
  loadB(0, 0);
  __syncthreads();

  const int KT = K >> 5;
  for (int kt = 0; kt < KT; kt++) {
    const int cur = kt & 1;
    if (kt + 1 < KT) {            // stream next tile into the other buffer
      loadA((kt + 1) * 32, cur ^ 1);
      loadB((kt + 1) * 32, cur ^ 1);
    }

    // A fragments: row = lane%16, k = (e<8?e:e+8) + 8*hi   (2x ds_load_b128)
    v16bf a[2], b[4];
    #pragma unroll
    for (int i = 0; i < 2; i++) {
      const __bf16* base = &sA[cur][(wr + i * 16 + rsel) * LDA + hi * 8];
      #pragma unroll
      for (int e = 0; e < 16; e++) a[i][e] = base[(e < 8) ? e : (e + 8)];
    }
    // B fragments from transposed tile: n = lane%16, k = (e<8?e:e+8) + 8*hi
    #pragma unroll
    for (int j = 0; j < 4; j++) {
      const __bf16* base = &sBt[cur][(wc + j * 16 + rsel) * LDBT + hi * 8];
      #pragma unroll
      for (int e = 0; e < 16; e++) b[j][e] = base[(e < 8) ? e : (e + 8)];
    }
    #pragma unroll
    for (int i = 0; i < 2; i++)
      #pragma unroll
      for (int j = 0; j < 4; j++)
        acc[i][j] = __builtin_amdgcn_wmma_f32_16x16x32_bf16(
            false, a[i], false, b[j], (short)0, acc[i][j], false, false);
    __syncthreads();
  }

  // epilogue (C layout: row = v + 8*hi, col = lane%16)
  #pragma unroll
  for (int i = 0; i < 2; i++) {
    #pragma unroll
    for (int j = 0; j < 4; j++) {
      #pragma unroll
      for (int v = 0; v < 8; v++) {
        const int row = m0 + wr + i * 16 + v + hi * 8;
        const int col = n0 + wc + j * 16 + rsel;
        float val = acc[i][j][v] + bias[col];
        if constexpr (EPI == EPI_PE) {
          const int t = row & (cfg::T - 1);
          const float dt = __expf((float)(col & ~1) * (-9.21034037f / 512.0f));
          val += (col & 1) ? __cosf((float)t * dt) : __sinf((float)t * dt);
        }
        if constexpr (EPI == EPI_RES) val += resid[(size_t)row * N + col];
        if constexpr (EPI == EPI_RELU) val = fmaxf(val, 0.0f);
        if constexpr (EPI == EPI_QKV || EPI == EPI_RELU)
          Cb[(size_t)row * N + col] = (__bf16)val;
        else
          Cf[(size_t)row * N + col] = val;
      }
    }
  }
}

// ---------------------------------------------------------------------------
// Flash attention per (b, h, 64-row q block).  128 threads = 4 waves,
// wave w owns rows w*16..w*16+15 of the q block (full softmax stays in-wave).
// K/V tiles streamed to LDS with GLOBAL_LOAD_ASYNC_TO_LDS_B128 when available.
// ---------------------------------------------------------------------------
__global__ __launch_bounds__(128) void attn_kernel(
    const __bf16* __restrict__ qkv, const int* __restrict__ obs,
    float* __restrict__ ctx)
{
  using namespace cfg;
  constexpr int LQ = 72, LK = 72, LV = 72, LP = 72;
  __shared__ __bf16 sQ[64 * LQ];
  __shared__ __bf16 sK[64 * LK];
  __shared__ __bf16 sV[64 * LV];
  __shared__ __bf16 sP[64 * LP];

  const int tid  = threadIdx.x;
  const int wv   = tid >> 5;
  const int lane = tid & 31;
  const int rsel = lane & 15;
  const int hi   = lane >> 4;
  const int qb = blockIdx.x, h = blockIdx.y, b = blockIdx.z;
  const int q0 = qb * 64;
  const int ob = obs[b];
  const size_t rowBase = (size_t)b * T;   // row index into qkv (stride 3*D)

  // load Q tile (64 x 64 bf16)
  {
    const int r = tid >> 1, c0 = (tid & 1) * 32;
    const __bf16* src = qkv + (rowBase + q0 + r) * (3 * D) + h * Dh + c0;
    #pragma unroll
    for (int c = 0; c < 32; c++) sQ[r * LQ + c0 + c] = src[c];
  }

  v8f o[4] = {};
  float mrow[8], lrow[8];
  #pragma unroll
  for (int v = 0; v < 8; v++) { mrow[v] = -3.0e38f; lrow[v] = 0.0f; }

  const int njb = qb + 1;   // causal: only key blocks j0 <= q0
  for (int jb = 0; jb < njb; jb++) {
    const int j0 = jb * 64;
    __syncthreads();
    {
      const int r = tid >> 1, c0 = (tid & 1) * 32;
      const __bf16* srck = qkv + (rowBase + j0 + r) * (3 * D) + D     + h * Dh + c0;
      const __bf16* srcv = qkv + (rowBase + j0 + r) * (3 * D) + 2 * D + h * Dh + c0;
#ifdef HAVE_ASYNC_LDS
      #pragma unroll
      for (int c = 0; c < 32; c += 8) {   // 16 bytes per async op per lane
        __builtin_amdgcn_global_load_async_to_lds_b128(
            (AS_GLOBAL v4i*)(srck + c),
            (AS_LDS v4i*)&sK[r * LK + c0 + c], 0, 0);
        __builtin_amdgcn_global_load_async_to_lds_b128(
            (AS_GLOBAL v4i*)(srcv + c),
            (AS_LDS v4i*)&sV[r * LV + c0 + c], 0, 0);
      }
      wait_asynccnt0();
#else
      #pragma unroll
      for (int c = 0; c < 32; c++) {
        sK[r * LK + c0 + c] = srck[c];
        sV[r * LV + c0 + c] = srcv[c];
      }
#endif
    }
    __syncthreads();

    // S = Q @ K^T  (64x64 per block; this wave: 16x64 -> 4 tiles)
    v8f s[4] = {};
    #pragma unroll
    for (int d0 = 0; d0 < Dh; d0 += 32) {
      v16bf qa;
      const __bf16* qbase = &sQ[(wv * 16 + rsel) * LQ + d0 + hi * 8];
      #pragma unroll
      for (int e = 0; e < 16; e++) qa[e] = qbase[(e < 8) ? e : (e + 8)];
      #pragma unroll
      for (int nt = 0; nt < 4; nt++) {
        v16bf kb;  // B[d][j] = K[j][d] -> contiguous read of sK row j
        const __bf16* kbase = &sK[(nt * 16 + rsel) * LK + d0 + hi * 8];
        #pragma unroll
        for (int e = 0; e < 16; e++) kb[e] = kbase[(e < 8) ? e : (e + 8)];
        s[nt] = __builtin_amdgcn_wmma_f32_16x16x32_bf16(
            false, qa, false, kb, (short)0, s[nt], false, false);
      }
    }

    // scale + masks (causal, context window, padding) exactly as reference
    float ps[4][8];
    #pragma unroll
    for (int nt = 0; nt < 4; nt++) {
      #pragma unroll
      for (int v = 0; v < 8; v++) {
        const int i = q0 + wv * 16 + v + hi * 8;
        const int j = j0 + nt * 16 + rsel;
        float sv = s[nt][v] * 0.125f;    // 1/sqrt(64)
        const bool win = (i > CTX) && (i < ob);
        if ((j > i) || (win && (j < i - CTX))) sv -= 1.0e9f;
        if (j > ob) sv -= 1.0e9f;
        ps[nt][v] = sv;
      }
    }

    // online softmax: row lives entirely in one lane-half -> width-16 shuffles
    #pragma unroll
    for (int v = 0; v < 8; v++) {
      float mn = mrow[v];
      #pragma unroll
      for (int nt = 0; nt < 4; nt++) mn = fmaxf(mn, ps[nt][v]);
      #pragma unroll
      for (int off = 1; off < 16; off <<= 1) mn = fmaxf(mn, __shfl_xor(mn, off, 16));
      const float alpha = __expf(mrow[v] - mn);
      mrow[v] = mn;
      float rs = 0.0f;
      #pragma unroll
      for (int nt = 0; nt < 4; nt++) {
        const float p = __expf(ps[nt][v] - mn);
        sP[(wv * 16 + v + hi * 8) * LP + nt * 16 + rsel] = (__bf16)p;
        rs += p;
      }
      #pragma unroll
      for (int off = 1; off < 16; off <<= 1) rs += __shfl_xor(rs, off, 16);
      lrow[v] = lrow[v] * alpha + rs;
      #pragma unroll
      for (int nd = 0; nd < 4; nd++) o[nd][v] *= alpha;
    }
    __syncthreads();

    // O += P @ V  (K dim = 64 key positions -> 2 wmma k-steps)
    #pragma unroll
    for (int ks = 0; ks < 2; ks++) {
      v16bf pa;
      const __bf16* pbase = &sP[(wv * 16 + rsel) * LP + ks * 32 + hi * 8];
      #pragma unroll
      for (int e = 0; e < 16; e++) pa[e] = pbase[(e < 8) ? e : (e + 8)];
      #pragma unroll
      for (int nd = 0; nd < 4; nd++) {
        v16bf vb;  // B[k][n] = V[k][n] -> strided column read
        const __bf16* vbase = &sV[(ks * 32 + hi * 8) * LV + nd * 16 + rsel];
        #pragma unroll
        for (int e = 0; e < 16; e++) vb[e] = vbase[((e < 8) ? e : (e + 8)) * LV];
        o[nd] = __builtin_amdgcn_wmma_f32_16x16x32_bf16(
            false, pa, false, vb, (short)0, o[nd], false, false);
      }
    }
  }

  // normalize and write ctx (f32, [B,T,D] with head offset)
  #pragma unroll
  for (int nd = 0; nd < 4; nd++) {
    #pragma unroll
    for (int v = 0; v < 8; v++) {
      const int t = q0 + wv * 16 + v + hi * 8;
      const int d = nd * 16 + rsel;
      ctx[((size_t)b * T + t) * D + h * Dh + d] = o[nd][v] / lrow[v];
    }
  }
}

// ---------------------------------------------------------------------------
// Row-wise LayerNorm over D=512 (one block / row, 256 threads).
// ---------------------------------------------------------------------------
__global__ __launch_bounds__(256) void ln_kernel(
    const float* __restrict__ z, const float* __restrict__ g,
    const float* __restrict__ bb, float* __restrict__ out)
{
  __shared__ float red[16];
  const int row = blockIdx.x;
  const int t = threadIdx.x;
  const float* p = z + (size_t)row * 512;
  const float a0 = p[t], a1 = p[t + 256];
  float s = a0 + a1, q = a0 * a0 + a1 * a1;
  #pragma unroll
  for (int off = 16; off > 0; off >>= 1) {
    s += __shfl_xor(s, off, 32);
    q += __shfl_xor(q, off, 32);
  }
  const int w = t >> 5;
  if ((t & 31) == 0) { red[w] = s; red[8 + w] = q; }
  __syncthreads();
  if (t == 0) {
    float S = 0.0f, Q = 0.0f;
    #pragma unroll
    for (int i = 0; i < 8; i++) { S += red[i]; Q += red[8 + i]; }
    red[0] = S; red[8] = Q;
  }
  __syncthreads();
  const float mu = red[0] * (1.0f / 512.0f);
  const float var = red[8] * (1.0f / 512.0f) - mu * mu;
  const float rs = rsqrtf(var + 1e-5f);
  out[(size_t)row * 512 + t]       = (a0 - mu) * rs * g[t]       + bb[t];
  out[(size_t)row * 512 + t + 256] = (a1 - mu) * rs * g[t + 256] + bb[t + 256];
}

// ---------------------------------------------------------------------------
// Masked mean over time: out[b,d] = sum_{t<=obs[b]} x[b,t,d] / (obs[b]+1)
// ---------------------------------------------------------------------------
__global__ __launch_bounds__(512) void agg_kernel(
    const float* __restrict__ x, const int* __restrict__ obs,
    float* __restrict__ out)
{
  using namespace cfg;
  const int b = blockIdx.x, c = threadIdx.x;
  const int ob = obs[b];
  const float* p = x + (size_t)b * T * D + c;
  float s = 0.0f;
  for (int t = 0; t <= ob; t++) s += p[(size_t)t * D];
  out[b * D + c] = s / (float)(ob + 1);
}

// ---------------------------------------------------------------------------
extern "C" void kernel_launch(void* const* d_in, const int* in_sizes, int n_in,
                              void* d_out, int out_size, void* d_ws, size_t ws_size,
                              hipStream_t stream)
{
  using namespace cfg;
  (void)in_sizes; (void)n_in; (void)out_size; (void)ws_size;

  const float* traj = (const float*)d_in[0];
  const int*   obs  = (const int*)d_in[1];
  const float* w_in = (const float*)d_in[2];
  const float* b_in = (const float*)d_in[3];

  char* ws = (char*)d_ws;
  const size_t SZF = (size_t)BT * D * sizeof(float);            // 64 MB
  float*  x    = (float*)(ws);
  float*  z    = (float*)(ws + SZF);
  float*  ctx  = (float*)(ws + 2 * SZF);
  __bf16* qkvb = (__bf16*)(ws + 3 * SZF);
  __bf16* hidb = (__bf16*)(ws + 3 * SZF + (size_t)BT * 3 * D * sizeof(__bf16));

  const dim3 blk(256);
  const int MB = BT / 128;   // 256 row tiles

  // x = traj @ w_in + b_in + PE
  gemm_bf16_kernel<EPI_PE, float><<<dim3(D / 128, MB), blk, 0, stream>>>(
      traj, w_in, b_in, nullptr, x, nullptr, BT, D, D);

  for (int l = 0; l < 4; l++) {
    const int base = 4 + 12 * l;
    const float* w_qkv = (const float*)d_in[base + 0];
    const float* b_qkv = (const float*)d_in[base + 1];
    const float* w_o   = (const float*)d_in[base + 2];
    const float* b_o   = (const float*)d_in[base + 3];
    const float* w_ff1 = (const float*)d_in[base + 4];
    const float* b_ff1 = (const float*)d_in[base + 5];
    const float* w_ff2 = (const float*)d_in[base + 6];
    const float* b_ff2 = (const float*)d_in[base + 7];
    const float* ln1_g = (const float*)d_in[base + 8];
    const float* ln1_b = (const float*)d_in[base + 9];
    const float* ln2_g = (const float*)d_in[base + 10];
    const float* ln2_b = (const float*)d_in[base + 11];

    // qkv(bf16) = x @ w_qkv + b_qkv
    gemm_bf16_kernel<EPI_QKV, float><<<dim3(3 * D / 128, MB), blk, 0, stream>>>(
        x, w_qkv, b_qkv, nullptr, nullptr, qkvb, BT, 3 * D, D);
    // ctx = flash attention
    attn_kernel<<<dim3(T / 64, H, B), dim3(128), 0, stream>>>(qkvb, obs, ctx);
    // z = x + ctx @ w_o + b_o
    gemm_bf16_kernel<EPI_RES, float><<<dim3(D / 128, MB), blk, 0, stream>>>(
        ctx, w_o, b_o, x, z, nullptr, BT, D, D);
    // x = LN1(z)
    ln_kernel<<<dim3(BT), blk, 0, stream>>>(z, ln1_g, ln1_b, x);
    // hid(bf16) = relu(x @ w_ff1 + b_ff1)
    gemm_bf16_kernel<EPI_RELU, float><<<dim3(FF / 128, MB), blk, 0, stream>>>(
        x, w_ff1, b_ff1, nullptr, nullptr, hidb, BT, FF, D);
    // z = x + hid @ w_ff2 + b_ff2
    gemm_bf16_kernel<EPI_RES, __bf16><<<dim3(D / 128, MB), blk, 0, stream>>>(
        hidb, w_ff2, b_ff2, x, z, nullptr, BT, D, FF);
    // x = LN2(z)
    ln_kernel<<<dim3(BT), blk, 0, stream>>>(z, ln2_g, ln2_b, x);
  }

  agg_kernel<<<dim3(B), dim3(512), 0, stream>>>(x, obs, (float*)d_out);
}
